// SignalMLPVQ_56684978373198
// MI455X (gfx1250) — matrix operand order repaired and checked by reading, not compiled, not run on hardware
//
#include <hip/hip_runtime.h>

// ---------------- types ----------------
typedef __attribute__((ext_vector_type(16))) __bf16 v16bf;
typedef __attribute__((ext_vector_type(8)))  float  v8f;

struct u4x2 { uint4 a, b; };

// ---------------- problem constants ----------------
constexpr int NB  = 16384;   // batch rows
constexpr int K1  = 4096;    // T*D
constexpr int LD  = 1024;    // latent dim
constexpr int NC  = 8192;    // codebook entries
constexpr int NO  = 128;     // output dim

__device__ __forceinline__ v16bf pack16(float4 a0, float4 a1, float4 a2,
                                        float4 a3) {
  v16bf a;
  a[0]  = (__bf16)a0.x; a[1]  = (__bf16)a0.y; a[2]  = (__bf16)a0.z; a[3]  = (__bf16)a0.w;
  a[4]  = (__bf16)a1.x; a[5]  = (__bf16)a1.y; a[6]  = (__bf16)a1.z; a[7]  = (__bf16)a1.w;
  a[8]  = (__bf16)a2.x; a[9]  = (__bf16)a2.y; a[10] = (__bf16)a2.z; a[11] = (__bf16)a2.w;
  a[12] = (__bf16)a3.x; a[13] = (__bf16)a3.y; a[14] = (__bf16)a3.z; a[15] = (__bf16)a3.w;
  return a;
}

__device__ __forceinline__ void wmma_bf16(v16bf a, v16bf b, v8f& acc) {
  acc = __builtin_amdgcn_wmma_f32_16x16x32_bf16(false, a, false, b, (short)0,
                                                acc, false, false);
}

// ---------------- tiny convert kernels ----------------
__global__ void k_cvt_bf16(const float* __restrict__ src,
                           __bf16* __restrict__ dst, int n) {
  int i = blockIdx.x * 256 + threadIdx.x;
  if (i < n) dst[i] = (__bf16)src[i];
}

// per-codebook-row: convert to bf16 and compute ||e||^2
__global__ void k_cb_prep(const float* __restrict__ cb,
                          __bf16* __restrict__ cbb,
                          float* __restrict__ enorm) {
  __shared__ float red[8];
  int row = blockIdx.x, t = threadIdx.x;
  const float4 v = *(const float4*)(cb + (long)row * LD + t * 4);
  __bf16* o = cbb + (long)row * LD + t * 4;
  o[0] = (__bf16)v.x; o[1] = (__bf16)v.y; o[2] = (__bf16)v.z; o[3] = (__bf16)v.w;
  float s = v.x * v.x + v.y * v.y + v.z * v.z + v.w * v.w;
  for (int m = 1; m < 32; m <<= 1) s += __shfl_xor(s, m, 32);
  if ((t & 31) == 0) red[t >> 5] = s;
  __syncthreads();
  if (t == 0) {
    float tot = 0.f;
    for (int i = 0; i < 8; i++) tot += red[i];
    enorm[row] = tot;
  }
}

// ---------------- encoder GEMM: z_e = x @ W_enc^T + b_enc ----------------
// One block per 16-row M-tile. The block converts its x rows to bf16 ONCE
// into 128KB dynamic LDS (WMMA A-fragment layout); 8 waves sweep all 64
// N-tiles, 4 at a time with 4 accumulators; A and 4 B fragments are
// double-buffered in registers (loads run one k-step ahead of the WMMAs).
__global__ void __launch_bounds__(256)
k_encoder(const float* __restrict__ x, const __bf16* __restrict__ wenc,
          const float* __restrict__ benc, float* __restrict__ zef,
          __bf16* __restrict__ zeb) {
  extern __shared__ uint4 ldsx[];  // 128KB = 128 ksteps * 32 lanes * 32B
  const int tid = threadIdx.x;
  const int m0 = blockIdx.x * 16;
  // cooperative stage + convert: entry e = (kstep, lane)
  for (int e = tid; e < 128 * 32; e += 256) {
    const int ks = e >> 5, ln = e & 31;
    const int h2 = ln >> 4, mr = ln & 15;
    const float* xr = x + (long)(m0 + mr) * K1 + ks * 32 + h2 * 16;
    float4 a0 = ((const float4*)xr)[0];
    float4 a1 = ((const float4*)xr)[1];
    float4 a2 = ((const float4*)(xr + 8))[0];
    float4 a3 = ((const float4*)(xr + 8))[1];
    u4x2 u = __builtin_bit_cast(u4x2, pack16(a0, a1, a2, a3));
    ldsx[e * 2 + 0] = u.a;
    ldsx[e * 2 + 1] = u.b;
  }
  __syncthreads();
  const int wave = tid >> 5, lane = tid & 31;
  const int hl = lane >> 4, mrow = lane & 15;
  for (int ng = 0; ng < 2; ng++) {  // 8 N-tiles per wave = 2 groups of 4
    const int nbase = wave * 8 + ng * 4;
    const __bf16* wr[4];
#pragma unroll
    for (int j = 0; j < 4; j++)
      wr[j] = wenc + (long)((nbase + j) * 16 + mrow) * K1;
    v8f acc[4] = {};
    // prologue: k-step 0 operands
    u4x2 ua;
    ua.a = ldsx[lane * 2 + 0];
    ua.b = ldsx[lane * 2 + 1];
    v16bf b[4];
#pragma unroll
    for (int j = 0; j < 4; j++) b[j] = *(const v16bf*)(wr[j] + hl * 16);
    for (int ks = 0; ks < 128; ks++) {
      const int kn = (ks + 1 < 128) ? ks + 1 : ks;  // branch-free tail
      u4x2 un;
      un.a = ldsx[(kn * 32 + lane) * 2 + 0];
      un.b = ldsx[(kn * 32 + lane) * 2 + 1];
      v16bf bn[4];
      const int offn = kn * 32 + hl * 16;
#pragma unroll
      for (int j = 0; j < 4; j++) bn[j] = *(const v16bf*)(wr[j] + offn);
      v16bf a = __builtin_bit_cast(v16bf, ua);
#pragma unroll
      for (int j = 0; j < 4; j++) wmma_bf16(a, b[j], acc[j]);
      ua = un;
#pragma unroll
      for (int j = 0; j < 4; j++) b[j] = bn[j];
    }
#pragma unroll
    for (int j = 0; j < 4; j++) {
      const int col = (nbase + j) * 16 + mrow;
      const float bias = benc[col];
      for (int r = 0; r < 8; r++) {
        int m = m0 + hl * 8 + r;  // C layout: M = r + hl*8, N = mrow
        float v = acc[j][r] + bias;
        zef[(long)m * LD + col] = v;
        zeb[(long)m * LD + col] = (__bf16)v;
      }
    }
  }
}

// ---------------- fused distance GEMM + argmin ----------------
// 4 waves/block (128KB dynamic LDS); each wave owns 16 rows (64 rows per
// block per codebook sweep -> halved L2 traffic), sweeps all 8192 codes 64
// at a time with 4 accumulators; A + 4 B fragments double-buffered.
__global__ void __launch_bounds__(128)
k_vq_argmin(const __bf16* __restrict__ zeb, const __bf16* __restrict__ cbb,
            const float* __restrict__ enorm, int* __restrict__ idxw,
            float* __restrict__ out_idx) {
  extern __shared__ uint4 lds[];  // 128KB: 4 waves * 32 ksteps * 32 lanes * 32B
  const int wave = threadIdx.x >> 5, lane = threadIdx.x & 31;
  const int hl = lane >> 4, mrow = lane & 15;
  const int m0 = (blockIdx.x * 4 + wave) * 16;
  const int lbase = wave * 2048;
  const __bf16* zr = zeb + (long)(m0 + mrow) * LD;
  for (int ks = 0; ks < 32; ks++) {
    int kb = ks * 32;
    lds[lbase + ks * 64 + lane * 2 + 0] = *(const uint4*)(zr + kb + hl * 16);
    lds[lbase + ks * 64 + lane * 2 + 1] = *(const uint4*)(zr + kb + 8 + hl * 16);
  }
  __syncthreads();
  float runMin[8];
  int runIdx[8];
  for (int r = 0; r < 8; r++) { runMin[r] = 3.4e38f; runIdx[r] = 0; }
  for (int nt = 0; nt < NC / 16; nt += 4) {
    const __bf16* br[4];
#pragma unroll
    for (int j = 0; j < 4; j++)
      br[j] = cbb + (long)((nt + j) * 16 + mrow) * LD;
    __builtin_prefetch(br[0] + 64 * LD, 0, 1);  // next 64-code group
    v8f acc[4] = {};
    u4x2 ua;
    ua.a = lds[lbase + lane * 2 + 0];
    ua.b = lds[lbase + lane * 2 + 1];
    v16bf b[4];
#pragma unroll
    for (int j = 0; j < 4; j++) b[j] = *(const v16bf*)(br[j] + hl * 16);
    for (int ks = 0; ks < 32; ks++) {
      const int kn = (ks + 1 < 32) ? ks + 1 : ks;  // branch-free tail
      u4x2 un;
      un.a = lds[lbase + kn * 64 + lane * 2 + 0];
      un.b = lds[lbase + kn * 64 + lane * 2 + 1];
      v16bf bn[4];
      const int offn = kn * 32 + hl * 16;
#pragma unroll
      for (int j = 0; j < 4; j++) bn[j] = *(const v16bf*)(br[j] + offn);
      v16bf a = __builtin_bit_cast(v16bf, ua);
#pragma unroll
      for (int j = 0; j < 4; j++) wmma_bf16(a, b[j], acc[j]);
      ua = un;
#pragma unroll
      for (int j = 0; j < 4; j++) b[j] = bn[j];
    }
#pragma unroll
    for (int j = 0; j < 4; j++) {
      const int code = (nt + j) * 16 + mrow;
      const float en = enorm[code];
      for (int r = 0; r < 8; r++) {
        float v = en - 2.0f * acc[j][r];  // ||e||^2 - 2 z.e (||z||^2 row-const)
        if (v < runMin[r]) { runMin[r] = v; runIdx[r] = code; }
      }
    }
  }
  // cross-lane min over the 16 columns each half-group saw (wave32 shuffles)
  for (int mask = 1; mask < 16; mask <<= 1) {
    for (int r = 0; r < 8; r++) {
      float ov = __shfl_xor(runMin[r], mask, 32);
      int oi = __shfl_xor(runIdx[r], mask, 32);
      if (ov < runMin[r] || (ov == runMin[r] && oi < runIdx[r])) {
        runMin[r] = ov; runIdx[r] = oi;
      }
    }
  }
  if (mrow == 0) {
    for (int r = 0; r < 8; r++) {
      int m = m0 + hl * 8 + r;
      idxw[m] = runIdx[r];
      out_idx[m] = (float)runIdx[r];
    }
  }
}

// ---------------- vq loss: deterministic two-stage reduction ----------------
__global__ void k_vq_loss(const float* __restrict__ zef,
                          const float* __restrict__ cb,
                          const int* __restrict__ idxw,
                          float* __restrict__ part) {
  __shared__ float red[8];
  int row = blockIdx.x, t = threadIdx.x;
  const float* q = cb + (long)idxw[row] * LD;
  const float* z = zef + (long)row * LD;
  float s = 0.f;
  for (int i = t * 4; i < t * 4 + 4; i++) {
    float d = q[i] - z[i];
    s += d * d;
  }
  for (int m = 1; m < 32; m <<= 1) s += __shfl_xor(s, m, 32);
  if ((t & 31) == 0) red[t >> 5] = s;
  __syncthreads();
  if (t == 0) {
    float tot = 0.f;
    for (int i = 0; i < 8; i++) tot += red[i];
    part[row] = tot;
  }
}

__global__ void k_loss_final(const float* __restrict__ part,
                             float* __restrict__ loss) {
  __shared__ float red[8];
  int t = threadIdx.x;
  float s = 0.f;
  for (int i = t; i < NB; i += 256) s += part[i];
  for (int m = 1; m < 32; m <<= 1) s += __shfl_xor(s, m, 32);
  if ((t & 31) == 0) red[t >> 5] = s;
  __syncthreads();
  if (t == 0) {
    float tot = 0.f;
    for (int i = 0; i < 8; i++) tot += red[i];
    // vq_loss = (1 + 0.25) * mean((quantized - z_e)^2)
    *loss = tot * (1.25f / ((float)NB * (float)LD));
  }
}

// ---------------- classifier GEMM + softmax ----------------
__global__ void __launch_bounds__(256)
k_classifier(const __bf16* __restrict__ cbb, const __bf16* __restrict__ wcls,
             const float* __restrict__ bcls, const int* __restrict__ idxw,
             float* __restrict__ out) {
  __shared__ float sl[16 * NO];
  const int wave = threadIdx.x >> 5, lane = threadIdx.x & 31;
  const int hl = lane >> 4, mrow = lane & 15;
  const int m0 = blockIdx.x * 16;
  const __bf16* ar = cbb + (long)idxw[m0 + mrow] * LD;
  const __bf16* br = wcls + (long)(wave * 16 + mrow) * LD;
  v8f acc = {};
  for (int kb = 0; kb < LD; kb += 32) {
    u4x2 u;
    u.a = *(const uint4*)(ar + kb + hl * 16);
    u.b = *(const uint4*)(ar + kb + 8 + hl * 16);
    v16bf a = __builtin_bit_cast(v16bf, u);
    v16bf b = *(const v16bf*)(br + kb + hl * 16);
    wmma_bf16(a, b, acc);
  }
  const int col = wave * 16 + mrow;
  const float bias = bcls[col];
  for (int r = 0; r < 8; r++) sl[(hl * 8 + r) * NO + col] = acc[r] + bias;
  __syncthreads();
  if (threadIdx.x < 16) {
    int r = threadIdx.x;
    float mx = -3.4e38f;
    for (int c = 0; c < NO; c++) mx = fmaxf(mx, sl[r * NO + c]);
    float sum = 0.f;
    for (int c = 0; c < NO; c++) {
      float e = __expf(sl[r * NO + c] - mx);
      sl[r * NO + c] = e;
      sum += e;
    }
    float inv = 1.0f / sum;
    float* o = out + (long)(m0 + r) * NO;
    for (int c = 0; c < NO; c++) o[c] = sl[r * NO + c] * inv;
  }
}

// ---------------- workspace layout (bytes) ----------------
constexpr size_t O_ZEF   = 0;                                 // z_e fp32   64 MB
constexpr size_t O_ZEB   = O_ZEF + (size_t)NB * LD * 4;       // z_e bf16   32 MB
constexpr size_t O_WENCB = O_ZEB + (size_t)NB * LD * 2;       // W_enc bf16  8 MB
constexpr size_t O_CBB   = O_WENCB + (size_t)LD * K1 * 2;     // cb bf16    16 MB
constexpr size_t O_WCLSB = O_CBB + (size_t)NC * LD * 2;       // W_cls bf16
constexpr size_t O_ENORM = O_WCLSB + (size_t)NO * LD * 2;     // ||e||^2
constexpr size_t O_IDX   = O_ENORM + (size_t)NC * 4;          // idx int32
constexpr size_t O_PART  = O_IDX + (size_t)NB * 4;            // loss partials

extern "C" void kernel_launch(void* const* d_in, const int* in_sizes, int n_in,
                              void* d_out, int out_size, void* d_ws,
                              size_t ws_size, hipStream_t stream) {
  const float* x    = (const float*)d_in[0];
  const float* Wenc = (const float*)d_in[1];
  const float* benc = (const float*)d_in[2];
  const float* cb   = (const float*)d_in[3];
  const float* Wcls = (const float*)d_in[4];
  const float* bcls = (const float*)d_in[5];

  char* ws = (char*)d_ws;
  float*  zef   = (float*)(ws + O_ZEF);
  __bf16* zeb   = (__bf16*)(ws + O_ZEB);
  __bf16* wenb  = (__bf16*)(ws + O_WENCB);
  __bf16* cbb   = (__bf16*)(ws + O_CBB);
  __bf16* wclb  = (__bf16*)(ws + O_WCLSB);
  float*  enorm = (float*)(ws + O_ENORM);
  int*    idxw  = (int*)(ws + O_IDX);
  float*  part  = (float*)(ws + O_PART);

  float* out        = (float*)d_out;
  float* out_logits = out;                        // [16384*128]
  float* out_loss   = out + (size_t)NB * NO;      // [1]
  float* out_idx    = out + (size_t)NB * NO + 1;  // [16384]

  k_cvt_bf16<<<(LD * K1 + 255) / 256, 256, 0, stream>>>(Wenc, wenb, LD * K1);
  k_cvt_bf16<<<(NO * LD + 255) / 256, 256, 0, stream>>>(Wcls, wclb, NO * LD);
  k_cb_prep<<<NC, 256, 0, stream>>>(cb, cbb, enorm);
  k_encoder<<<NB / 16, 256, 131072, stream>>>(x, wenb, benc, zef, zeb);
  k_vq_argmin<<<NB / 64, 128, 131072, stream>>>(zeb, cbb, enorm, idxw, out_idx);
  k_vq_loss<<<NB, 256, 0, stream>>>(zef, cb, idxw, part);
  k_loss_final<<<1, 256, 0, stream>>>(part, out_loss);
  k_classifier<<<NB / 16, 256, 0, stream>>>(cbb, wclb, bcls, idxw, out_logits);
}
